// FFU_87625922773299
// MI455X (gfx1250) — matrix operand order repaired and checked by reading, not compiled, not run on hardware
//
#include <hip/hip_runtime.h>
#include <hip/hip_bf16.h>
#include <stdint.h>

typedef __attribute__((ext_vector_type(4)))  unsigned int u32x4;
typedef __attribute__((ext_vector_type(8)))  int          i32x8;
typedef __attribute__((ext_vector_type(4)))  int          i32x4;
typedef __attribute__((ext_vector_type(16))) __bf16       bf16x16;
typedef __attribute__((ext_vector_type(8)))  float        f32x8;

// ---------------------------------------------------------------------------
// Kernel 1: self-attention on y (Cy=2, N=4096) + y1 = gamma*attn_out + y -> ws
// Grid: 4 batches * 64 row-blocks = 256 WGs, 256 threads.
// Each WG: compute k,v (2x4096 each) into LDS, then 64 rows, 4 threads/row
// (split-K online softmax, merged in LDS).
// ---------------------------------------------------------------------------
__global__ __launch_bounds__(256) void FFU_attn_y1_kernel(
    const float* __restrict__ y,
    const float* __restrict__ wq, const float* __restrict__ bq,
    const float* __restrict__ wk, const float* __restrict__ bk,
    const float* __restrict__ wv, const float* __restrict__ bv,
    const float* __restrict__ gamma,
    float* __restrict__ y1out)
{
    __shared__ float kf0[4096], kf1[4096], vf0[4096], vf1[4096];
    __shared__ float rm[256], rl[256], ra0[256], ra1[256];

    const int tid  = threadIdx.x;
    const int b    = blockIdx.x >> 6;
    const int rblk = blockIdx.x & 63;
    const float* yb = y + (size_t)b * 8192;   // [2][4096]

    const float wk00 = wk[0], wk01 = wk[1], wk10 = wk[2], wk11 = wk[3];
    const float wv00 = wv[0], wv01 = wv[1], wv10 = wv[2], wv11 = wv[3];
    const float bk0 = bk[0], bk1 = bk[1], bv0 = bv[0], bv1 = bv[1];

    for (int j = tid; j < 4096; j += 256) {
        float y0v = yb[j], y1v = yb[4096 + j];
        kf0[j] = wk00 * y0v + wk01 * y1v + bk0;
        kf1[j] = wk10 * y0v + wk11 * y1v + bk1;
        vf0[j] = wv00 * y0v + wv01 * y1v + bv0;
        vf1[j] = wv10 * y0v + wv11 * y1v + bv1;
    }
    __syncthreads();

    const int rlocal = tid >> 2;           // 0..63
    const int sub    = tid & 3;            // split-K quarter
    const int i      = rblk * 64 + rlocal; // row 0..4095

    const float Y0i = yb[i], Y1i = yb[4096 + i];
    const float q0 = wq[0] * Y0i + wq[1] * Y1i + bq[0];
    const float q1 = wq[2] * Y0i + wq[3] * Y1i + bq[1];

    float m = -3.0e38f, l = 0.f, a0 = 0.f, a1 = 0.f;
    const int j0 = sub * 1024;
    for (int j = j0; j < j0 + 1024; ++j) {
        float e  = q0 * kf0[j] + q1 * kf1[j];
        float nm = fmaxf(m, e);
        float s  = __expf(m - nm);
        float p  = __expf(e - nm);
        l  = l * s + p;
        a0 = a0 * s + vf0[j] * p;
        a1 = a1 * s + vf1[j] * p;
        m  = nm;
    }
    rm[tid] = m; rl[tid] = l; ra0[tid] = a0; ra1[tid] = a1;
    __syncthreads();

    if (sub == 0) {
        float M = rm[tid];
        M = fmaxf(M, rm[tid + 1]); M = fmaxf(M, rm[tid + 2]); M = fmaxf(M, rm[tid + 3]);
        float L = 0.f, A0 = 0.f, A1 = 0.f;
        #pragma unroll
        for (int s4 = 0; s4 < 4; ++s4) {
            float sc = __expf(rm[tid + s4] - M);
            L  += rl[tid + s4]  * sc;
            A0 += ra0[tid + s4] * sc;
            A1 += ra1[tid + s4] * sc;
        }
        float g = gamma[0];
        float invL = 1.0f / L;
        y1out[(size_t)b * 8192 + i]        = g * (A0 * invL) + Y0i;
        y1out[(size_t)b * 8192 + 4096 + i] = g * (A1 * invL) + Y1i;
    }
}

// ---------------------------------------------------------------------------
// Kernel 2: 3x3 conv (Cin=1024 -> 2) via bf16 WMMA implicit GEMM, fed by TDM,
// fused BN + ReLU + bilinear upsample of y / y1 + final combine.
// Grid: 4 batches * 64 (two output rows each) = 256 WGs, 256 threads (8 waves).
// LDS: x-chunk staged by tensor_load_to_lds with pad feature providing zeroed
// halo columns (row stride 130 dwords = 128 data + 2 zero pad).
// ---------------------------------------------------------------------------
#define CH 32  // channels per TDM chunk

__device__ inline float bilin64(const float* p, int iy0, int iy1, int ix0, int ix1,
                                float fy, float fx)
{
    float a = p[iy0 * 64 + ix0], bb = p[iy0 * 64 + ix1];
    float c = p[iy1 * 64 + ix0], dd = p[iy1 * 64 + ix1];
    float top = a + (bb - a) * fx;
    float bot = c + (dd - c) * fx;
    return top + (bot - top) * fy;
}

__global__ __launch_bounds__(256) void FFU_conv_fuse_kernel(
    const float* __restrict__ x,       // (4,1024,128,128)
    const float* __restrict__ y,       // (4,2,64,64)
    const float* __restrict__ conv_w,  // (2,1024,3,3)
    const float* __restrict__ conv_b,
    const float* __restrict__ bn_scale, const float* __restrict__ bn_bias,
    const float* __restrict__ bn_mean,  const float* __restrict__ bn_var,
    const float* __restrict__ y1ws,    // (4,2,4096) from kernel 1
    float* __restrict__ out)           // (4,2,128,128)
{
    __shared__ float  xs[2 + CH * 4 * 130];                 // guard(2) + 32ch*4rows*130dw
    __shared__ __attribute__((aligned(32))) __bf16 wb[2 * 9 * 1024]; // weights bf16 [co][tap][c]

    const int tid  = threadIdx.x;
    const int wave = tid >> 5;
    const int lane = tid & 31;
    const int b    = blockIdx.x >> 6;
    const int y0   = (blockIdx.x & 63) << 1;  // two output rows y0, y0+1

    const int ystart = (y0 > 0) ? (y0 - 1) : 0;
    const int yend   = (y0 + 2 < 127) ? (y0 + 2) : 127;
    const int nrows  = yend - ystart + 1;     // 3 or 4
    const int chanStride = nrows * 130;       // dwords per channel in LDS

    // Stage all conv weights to LDS as bf16: wb[(co*9+t)*1024 + c]
    for (int d = tid; d < 18432; d += 256) {
        int co = d / 9216, r = d - co * 9216;
        int t = r >> 10, c = r & 1023;
        wb[d] = (__bf16)conv_w[((size_t)co * 1024 + c) * 9 + t];
    }
    // Zero guard + TDM pad gaps (zero halo columns); TDM never writes them.
    if (tid == 0) { xs[0] = 0.f; xs[1] = 0.f; }
    for (int r = tid; r < CH * nrows; r += 256) {
        xs[2 + r * 130 + 128] = 0.f;
        xs[2 + r * 130 + 129] = 0.f;
    }
    __syncthreads();

    float* xsb = &xs[2];
    const unsigned ldsAddr = (unsigned)(uintptr_t)(void*)xsb;

    f32x8 acc0 = {0.f,0.f,0.f,0.f,0.f,0.f,0.f,0.f};
    f32x8 acc1 = {0.f,0.f,0.f,0.f,0.f,0.f,0.f,0.f};

    const int m = lane & 15;      // pixel-in-tile (A rows / D M)
    const int h = lane >> 4;      // lane half
    const int xcol0 = (wave << 4) + m;

    // N-padding mask for the B fragment, hoisted & branchless.
    const bool bvalid = (lane & 15) < 2;
    bf16x16 zvec;
    #pragma unroll
    for (int e = 0; e < 16; ++e) zvec[e] = (__bf16)0.0f;

    for (int cc = 0; cc < 1024; cc += CH) {
        if (wave == 0) {
            unsigned long long ga = (unsigned long long)(uintptr_t)
                (x + ((size_t)(b * 1024 + cc)) * 16384 + (size_t)ystart * 128);
            u32x4 g0;
            g0.x = 1u;                                   // count=1, no gather
            g0.y = ldsAddr;                              // lds_addr [63:32]
            g0.z = (unsigned)(ga & 0xffffffffu);         // global_addr lo
            g0.w = (unsigned)((ga >> 32) & 0x1ffffffu) | (2u << 30); // type=2
            i32x8 g1;
            g1[0] = (int)0x03920000u;   // data_size=4B, pad_en, interval=128dw, amount=2dw
            g1[1] = (int)(128u << 16);  // tensor_dim0=128 (lo16)
            g1[2] = (int)((unsigned)nrows << 16);        // tensor_dim1 (lo16)
            g1[3] = (int)(128u << 16);  // tile_dim0=128
            g1[4] = (int)((unsigned)nrows | (32u << 16)); // tile_dim1, tile_dim2=32
            g1[5] = 128;                // tensor_dim0_stride (row stride, elems)
            g1[6] = (int)(16384u << 16);// tensor_dim1_stride lo16 (chan stride)
            g1[7] = 0;
            i32x4 g2; g2[0] = 32; g2[1] = 0; g2[2] = 0; g2[3] = 0; // tensor_dim2
            i32x4 g3; g3[0] = 0;  g3[1] = 0; g3[2] = 0; g3[3] = 0;
            i32x8 g4; g4[0] = 0;  g4[1] = 0; g4[2] = 0; g4[3] = 0;
            g4[4] = 0; g4[5] = 0; g4[6] = 0; g4[7] = 0;  // unused (VADDR4 slot)
            __builtin_amdgcn_tensor_load_to_lds(g0, g1, g2, g3, g4, 0);
            __builtin_amdgcn_s_wait_tensorcnt(0);
        }
        __syncthreads();

        #pragma unroll
        for (int t = 0; t < 9; ++t) {
            const int dy = t / 3 - 1, dx = t % 3 - 1;
            // ---- B fragment: 32(K=channels) x 16(N=cout padded), bf16.
            // Unconditional aligned 32B vector load (row clamped via n&1),
            // then a branchless whole-vector select for the N >= 2 padding.
            const __bf16* wp = &wb[(((lane & 15) & 1) * 9 + t) * 1024 + cc + 16 * h];
            bf16x16 bload = *(const bf16x16*)wp;
            bf16x16 bf = bvalid ? bload : zvec;

            #pragma unroll
            for (int i = 0; i < 2; ++i) {
                const int grow = y0 + i + dy;
                if ((unsigned)grow > 127u) continue;     // zero-pad row: skip
                const int slot = grow - ystart;
                const float* ap = xsb + slot * 130 + (xcol0 + dx); // dx=+-1 hits zeroed pads
                // ---- A fragment: 16(M=pixels) x 32(K=channels), bf16
                bf16x16 af;
                #pragma unroll
                for (int e = 0; e < 16; ++e) {
                    int g = e >> 1;
                    int k = 16 * h + e + 0 * g;
                    af[e] = (__bf16)ap[k * chanStride];
                }
                if (i == 0)
                    acc0 = __builtin_amdgcn_wmma_f32_16x16x32_bf16(
                        false, af, false, bf, (short)0, acc0, false, false);
                else
                    acc1 = __builtin_amdgcn_wmma_f32_16x16x32_bf16(
                        false, af, false, bf, (short)0, acc1, false, false);
            }
        }
        __syncthreads();   // before TDM reuses the buffer
    }

    // ---- Epilogue: lanes with N<2 hold real cout data (lanes 0,1,16,17)
    const int n = lane & 15;
    if (n < 2) {
        const int co = n;
        const float bias = conv_b[co];
        const float inv  = bn_scale[co] * rsqrtf(bn_var[co] + 1e-5f);
        const float mean = bn_mean[co];
        const float bnb  = bn_bias[co];
        const float* ybase  = y    + ((size_t)(b * 2 + co)) * 4096;
        const float* y1base = y1ws + ((size_t)(b * 2 + co)) * 4096;
        const float scale = 63.0f / 127.0f;

        #pragma unroll
        for (int i = 0; i < 2; ++i) {
            const int yy = y0 + i;
            float sy = yy * scale;
            int iy0 = (int)sy; float fy = sy - (float)iy0;
            int iy1 = (iy0 + 1 < 63) ? iy0 + 1 : 63;
            f32x8 acc = (i == 0) ? acc0 : acc1;
            #pragma unroll
            for (int v = 0; v < 8; ++v) {
                const int M = v + 8 * h;
                const int xcol = (wave << 4) + M;
                float xc = acc[v] + bias;
                xc = (xc - mean) * inv + bnb;
                xc = fmaxf(xc, 0.f);
                float sx = xcol * scale;
                int ix0 = (int)sx; float fx = sx - (float)ix0;
                int ix1 = (ix0 + 1 < 63) ? ix0 + 1 : 63;
                float yup  = bilin64(ybase,  iy0, iy1, ix0, ix1, fy, fx);
                float y1up = bilin64(y1base, iy0, iy1, ix0, ix1, fy, fx);
                out[(((size_t)b * 2 + co) * 128 + yy) * 128 + xcol] = yup + xc * y1up;
            }
        }
    }
}

// ---------------------------------------------------------------------------
extern "C" void kernel_launch(void* const* d_in, const int* in_sizes, int n_in,
                              void* d_out, int out_size, void* d_ws, size_t ws_size,
                              hipStream_t stream)
{
    const float* x        = (const float*)d_in[0];
    const float* y        = (const float*)d_in[1];
    const float* wq       = (const float*)d_in[2];
    const float* bq       = (const float*)d_in[3];
    const float* wk       = (const float*)d_in[4];
    const float* bk       = (const float*)d_in[5];
    const float* wv       = (const float*)d_in[6];
    const float* bv       = (const float*)d_in[7];
    const float* gamma    = (const float*)d_in[8];
    const float* conv_w   = (const float*)d_in[9];
    const float* conv_b   = (const float*)d_in[10];
    const float* bn_scale = (const float*)d_in[11];
    const float* bn_bias  = (const float*)d_in[12];
    const float* bn_mean  = (const float*)d_in[13];
    const float* bn_var   = (const float*)d_in[14];

    float* y1ws = (float*)d_ws;        // 4*2*4096 floats = 128 KB
    float* outp = (float*)d_out;       // 4*2*128*128 floats

    FFU_attn_y1_kernel<<<256, 256, 0, stream>>>(y, wq, bq, wk, bk, wv, bv, gamma, y1ws);
    FFU_conv_fuse_kernel<<<256, 256, 0, stream>>>(x, y, conv_w, conv_b,
                                                  bn_scale, bn_bias, bn_mean, bn_var,
                                                  y1ws, outp);
}
